// Additive_flow_69518340653348
// MI455X (gfx1250) — matrix operand order repaired and checked by reading, not compiled, not run on hardware
//
#include <hip/hip_runtime.h>
#include <hip/hip_bf16.h>
#include <stdint.h>

// ---------------------------------------------------------------------------
// Additive coupling flow on MI455X (gfx1250).
//  * prep_kernel: f32 weights -> bf16, K-pair-interleaved + bank-swizzled,
//    into d_ws (one-time layout pass, ~10.6 MB, L2-resident afterwards).
//  * flow_kernel: z tile (128x64 f32) resident in LDS for all 65 steps.
//    Per step the Tensor Data Mover DMAs weights into LDS (TENSORcnt):
//      - W2(s) streams during layer-1 WMMAs
//      - W1(s+1) streams into a double buffer during layer-2 WMMAs
//    MLP = v_wmma_f32_16x16x32_bf16; scalar head via VALU + shfl_xor.
// LDS map (dynamic, 299008 B of the 320 KB WGP LDS):
//   [      0,  34816) z    : 128 rows * 68 dwords (padded, bank-safe)
//   [  34816, 102400) h    : 8 waves * 16 rows * 264 bf16 (padded)
//   [ 102400, 135168) W1p0 : 32 k-pair lines * 256 dwords (swizzled)
//   [ 135168, 167936) W1p1 : double buffer
//   [ 167936, 299008) W2p  : 128 k-pair lines * 256 dwords (swizzled)
// ---------------------------------------------------------------------------

typedef __attribute__((ext_vector_type(16))) __bf16        v16bf;
typedef __attribute__((ext_vector_type(2)))  __bf16        v2bf;
typedef __attribute__((ext_vector_type(8)))  float         v8f;
typedef __attribute__((ext_vector_type(4)))  unsigned int  u32x4;
typedef __attribute__((ext_vector_type(8)))  int           i32x8;
typedef __attribute__((ext_vector_type(4)))  int           i32x4;

#define STEPS 65
#define NCOLS 64
#define HID   256
#define RPW   16
#define RPB   128
#define ZSTR  68        // z row stride in dwords (64 + 4 pad -> 4-bank lane skew)
#define HSTR  264       // h row stride in bf16  (256 + 8 pad)

#define ZOFF   0
#define HOFF   34816
#define W1OFF0 102400
#define W1OFF1 135168
#define W2OFF  167936
#define SHMEM  299008

#define NW1   (STEPS * 32 * HID)     // dwords of packed W1 in ws
#define NW2   (STEPS * 128 * HID)    // dwords of packed W2 in ws

// f32 -> bf16 via native conversions (backend picks v_cvt_pk_bf16_f32 if present)
__device__ __forceinline__ uint32_t pack2(float lo, float hi) {
    v2bf p = { (__bf16)lo, (__bf16)hi };
    return __builtin_bit_cast(uint32_t, p);
}
__device__ __forceinline__ unsigned short bf1(float f) {
    __bf16 b = (__bf16)f;
    unsigned short u; __builtin_memcpy(&u, &b, 2); return u;
}
__device__ __forceinline__ v16bf as16(const uint32_t* u) {
    v16bf r; __builtin_memcpy(&r, u, 32); return r;
}
// bank swizzle for weight lines: lanes of one B-fragment hit 32 distinct banks
__device__ __forceinline__ int swz(int kl) { return ((kl ^ (kl >> 2)) & 3) << 4; }

// Issue one 2-D TDM copy: ndw contiguous dwords global -> LDS byte offset.
__device__ __forceinline__ void tdm_load(uint32_t lds_off, const void* gsrc, uint32_t ndw) {
    uint64_t ga = (uint64_t)(uintptr_t)gsrc;
    u32x4 g0; i32x8 g1; i32x4 gz = (i32x4)0;
    g0[0] = 1u;                                   // count=1, user descriptor
    g0[1] = lds_off;                              // D#.lds_addr
    g0[2] = (uint32_t)ga;                         // D#.global_addr[31:0]
    g0[3] = (uint32_t)(ga >> 32) | (2u << 30);    // addr[56:32] | type=2
    g1[0] = (int)(2u << 16);                      // data_size = 4 bytes
    g1[1] = (int)((ndw & 0xFFFFu) << 16);         // tensor_dim0[15:0]
    g1[2] = (int)((ndw >> 16) | (1u << 16));      // tensor_dim0[31:16] | tensor_dim1=1
    g1[3] = (int)((ndw & 0xFFFFu) << 16);         // tile_dim0 = ndw (<=65535)
    g1[4] = 1;                                    // tile_dim1 = 1
    g1[5] = (int)ndw;                             // tensor_dim0_stride
    g1[6] = 0;
    g1[7] = 0;
#if defined(__clang_major__) && __clang_major__ >= 23
    __builtin_amdgcn_tensor_load_to_lds(g0, g1, gz, gz, (i32x8)0, 0);
#else
    __builtin_amdgcn_tensor_load_to_lds(g0, g1, gz, gz, 0);
#endif
}

// ---------------- one-time weight repack: f32 -> bf16 pairs ----------------
__global__ void __launch_bounds__(256)
prep_kernel(const float* __restrict__ W1, const float* __restrict__ W2,
            uint32_t* __restrict__ ws) {
    int d = blockIdx.x * 256 + threadIdx.x;
    if (d < NW1) {                                  // W1: K padded 63 -> 64
        int st = d / (32 * HID), r = d % (32 * HID);
        int kl = r >> 8, col = r & 255, k = kl * 2;
        const float* Ws = W1 + (size_t)st * 63 * HID;
        float a0 = (k     < 63) ? Ws[(size_t)k * HID + col]       : 0.f;
        float a1 = (k + 1 < 63) ? Ws[(size_t)(k + 1) * HID + col] : 0.f;
        ws[(size_t)st * (32 * HID) + kl * 256 + (col ^ swz(kl))] = pack2(a0, a1);
    } else if (d < NW1 + NW2) {                     // W2
        int e = d - NW1;
        int st = e / (128 * HID), r = e % (128 * HID);
        int kl = r >> 8, col = r & 255, k = kl * 2;
        const float* Ws = W2 + (size_t)st * HID * HID;
        ws[(size_t)NW1 + (size_t)st * (128 * HID) + kl * 256 + (col ^ swz(kl))] =
            pack2(Ws[(size_t)k * HID + col], Ws[(size_t)(k + 1) * HID + col]);
    }
}

// ------------------------------- flow ---------------------------------------
__global__ void __launch_bounds__(256, 1)
flow_kernel(const float* __restrict__ x,  const float* __restrict__ sv,
            const uint32_t* __restrict__ w1g, const uint32_t* __restrict__ w2g,
            const float* __restrict__ b1, const float* __restrict__ b2,
            const float* __restrict__ W3, const float* __restrict__ b3,
            const int*   __restrict__ idx, float* __restrict__ out) {
    extern __shared__ char smem[];
    float*          zb  = (float*)(smem + ZOFF);
    unsigned short* hb  = (unsigned short*)(smem + HOFF);
    uint32_t*       w2p = (uint32_t*)(smem + W2OFF);

    const int tid  = threadIdx.x;
    const int wave = tid >> 5;
    const int lane = tid & 31;
    const int hl   = lane >> 4;        // half-wave
    const int mq   = lane & 15;        // M (A) / N (B,C,D) lane index
    const int row0 = wave * RPW;
    unsigned short* hw = hb + wave * RPW * HSTR;

    const size_t g0 = (size_t)blockIdx.x * RPB * NCOLS;

    for (int t = tid; t < RPB * NCOLS; t += 256)
        zb[(t >> 6) * ZSTR + (t & 63)] = x[g0 + t];

    // preload step 0's W1 into buffer 0 (streams while z tile loads)
    if (wave == 0) tdm_load(W1OFF0, w1g, 32 * HID);

    #pragma unroll 1
    for (int st = 0; st < STEPS; ++st) {
        __syncthreads();                 // prior step done; W2/W1[st&1] LDS free

        const int    ci  = idx[st];
        const float* b1g = b1 + (size_t)st * HID;
        const float* b2g = b2 + (size_t)st * HID;
        const float* W3g = W3 + (size_t)st * HID;
        const float* b3g = b3 + (size_t)st;
        uint32_t* w1p = (uint32_t*)(smem + ((st & 1) ? W1OFF1 : W1OFF0));

        // -- TDM: DMA this step's W2 (W1[st] was issued during step st-1) --
        if (wave == 0) tdm_load(W2OFF, w2g + (size_t)st * 128 * HID, 128 * HID);

        // -- coupling swap: z[:,ci] <-> z[:,63] ----------------------------
        if (tid < RPB) {
            float a = zb[tid * ZSTR + ci];
            zb[tid * ZSTR + ci] = zb[tid * ZSTR + 63];
            zb[tid * ZSTR + 63] = a;
        }
        if (st + 1 < STEPS)              // warm L2 with next step's W2
            __builtin_prefetch((const char*)(w2g + (size_t)(st + 1) * 128 * HID) + tid * 512, 0, 0);

        if (wave == 0) __builtin_amdgcn_s_wait_tensorcnt((short)1);  // W1 landed
        __syncthreads();                 // W1 + swap visible; W2 still in flight

        // ================= Layer 1: h = relu(z1 @ W1 + b1) ================
        uint32_t a1f[2][8];
        #pragma unroll
        for (int kt = 0; kt < 2; ++kt)
            #pragma unroll
            for (int v = 0; v < 8; ++v) {
                int kb = kt * 32 + hl * 8 + ((v < 4) ? 2 * v : 16 + 2 * (v - 4));
                const float* zr = &zb[(row0 + mq) * ZSTR + kb];
                a1f[kt][v] = pack2(zr[0], zr[1]);
            }
        #pragma unroll 2
        for (int nt = 0; nt < 16; ++nt) {
            float bias = b1g[nt * 16 + mq];
            v8f acc;
            #pragma unroll
            for (int r = 0; r < 8; ++r) acc[r] = bias;
            #pragma unroll
            for (int kt = 0; kt < 2; ++kt) {
                uint32_t bf[8];
                #pragma unroll
                for (int v = 0; v < 8; ++v) {
                    int kl = kt * 16 + hl * 4 + v + ((v >= 4) ? 4 : 0);
                    bf[v] = w1p[kl * 256 + ((nt * 16 + mq) ^ swz(kl))];
                }
                acc = __builtin_amdgcn_wmma_f32_16x16x32_bf16(
                        false, as16(a1f[kt]), false, as16(bf),
                        (short)0, acc, false, false);
            }
            #pragma unroll
            for (int r = 0; r < 8; ++r)
                hw[(r + 8 * hl) * HSTR + nt * 16 + mq] = bf1(fmaxf(acc[r], 0.0f));
        }

        if (wave == 0) {
            __builtin_amdgcn_s_wait_tensorcnt((short)0);             // W2 landed
            if (st + 1 < STEPS)          // next W1 streams under layer-2 WMMAs
                tdm_load((st & 1) ? W1OFF0 : W1OFF1,
                         w1g + (size_t)(st + 1) * 32 * HID, 32 * HID);
        }
        __syncthreads();

        // ============ Layer 2 + head: t = relu(h @ W2 + b2) @ w3 ==========
        uint32_t a2f[8][8];
        #pragma unroll
        for (int kt = 0; kt < 8; ++kt) {
            const uint4 q0 = *(const uint4*)(hw + mq * HSTR + kt * 32 + hl * 8);
            const uint4 q1 = *(const uint4*)(hw + mq * HSTR + kt * 32 + hl * 8 + 16);
            a2f[kt][0] = q0.x; a2f[kt][1] = q0.y; a2f[kt][2] = q0.z; a2f[kt][3] = q0.w;
            a2f[kt][4] = q1.x; a2f[kt][5] = q1.y; a2f[kt][6] = q1.z; a2f[kt][7] = q1.w;
        }
        float tp[8];
        #pragma unroll
        for (int r = 0; r < 8; ++r) tp[r] = 0.f;

        #pragma unroll 2
        for (int nt = 0; nt < 16; ++nt) {
            float bias = b2g[nt * 16 + mq];
            v8f acc;
            #pragma unroll
            for (int r = 0; r < 8; ++r) acc[r] = bias;
            #pragma unroll
            for (int kt = 0; kt < 8; ++kt) {
                uint32_t bf[8];
                #pragma unroll
                for (int v = 0; v < 8; ++v) {
                    int kl = kt * 16 + hl * 4 + v + ((v >= 4) ? 4 : 0);
                    bf[v] = w2p[kl * 256 + ((nt * 16 + mq) ^ swz(kl))];
                }
                acc = __builtin_amdgcn_wmma_f32_16x16x32_bf16(
                        false, as16(a2f[kt]), false, as16(bf),
                        (short)0, acc, false, false);
            }
            float w3v = W3g[nt * 16 + mq];
            #pragma unroll
            for (int r = 0; r < 8; ++r) tp[r] += fmaxf(acc[r], 0.0f) * w3v;
        }
        #pragma unroll
        for (int off = 1; off < 16; off <<= 1)
            #pragma unroll
            for (int r = 0; r < 8; ++r) tp[r] += __shfl_xor(tp[r], off, 32);

        if (mq == 0) {   // lane 0 -> rows +0..7, lane 16 -> rows +8..15
            float b3v = b3g[0];
            #pragma unroll
            for (int r = 0; r < 8; ++r)
                zb[(row0 + r + 8 * hl) * ZSTR + 63] += tp[r] + b3v;
        }
        __syncthreads();
        if (tid < RPB) {                 // swap back
            float a = zb[tid * ZSTR + ci];
            zb[tid * ZSTR + ci] = zb[tid * ZSTR + 63];
            zb[tid * ZSTR + 63] = a;
        }
    }

    __syncthreads();
    for (int t = tid; t < RPB * NCOLS; t += 256)
        out[g0 + t] = expf(sv[t & (NCOLS - 1)]) * zb[(t >> 6) * ZSTR + (t & 63)];
}

extern "C" void kernel_launch(void* const* d_in, const int* in_sizes, int n_in,
                              void* d_out, int out_size, void* d_ws, size_t ws_size,
                              hipStream_t stream) {
    (void)n_in; (void)ws_size; (void)out_size;
    const float* x  = (const float*)d_in[0];
    const float* sv = (const float*)d_in[1];
    const float* W1 = (const float*)d_in[2];
    const float* b1 = (const float*)d_in[3];
    const float* W2 = (const float*)d_in[4];
    const float* b2 = (const float*)d_in[5];
    const float* W3 = (const float*)d_in[6];
    const float* b3 = (const float*)d_in[7];
    const int*   ix = (const int*)d_in[8];
    float*    out = (float*)d_out;
    uint32_t* ws  = (uint32_t*)d_ws;

    const int B = in_sizes[0] / NCOLS;            // 131072
    const int blocks = B / RPB;                   // 1024 workgroups

    static_assert(sizeof(v16bf) == 32, "v16bf must span 8 VGPRs");

    prep_kernel<<<(NW1 + NW2 + 255) / 256, 256, 0, stream>>>(W1, W2, ws);

    hipFuncSetAttribute((const void*)flow_kernel,
                        hipFuncAttributeMaxDynamicSharedMemorySize, SHMEM);
    flow_kernel<<<blocks, 256, SHMEM, stream>>>(
        x, sv, ws, ws + NW1, b1, b2, W3, b3, ix, out);
}
